// MultiHeadAttention_63221918597146
// MI455X (gfx1250) — matrix operand order repaired and checked by reading, not compiled, not run on hardware
//
#include <hip/hip_runtime.h>
#include <math.h>

#define DM    512
#define DK    64
#define SLEN  2048
#define NHB   16          // N_HEAD * BATCH
#define MROWS 4096        // BATCH * SLEN

typedef __attribute__((ext_vector_type(16))) __bf16 v16bf;
typedef __attribute__((ext_vector_type(8)))  __bf16 v8bf;
typedef __attribute__((ext_vector_type(8)))  float  v8f;

struct FragPair { v16bf hi; v16bf lo; };

// ---------------------------------------------------------------- helpers --

static __device__ __forceinline__ void split_store(float a, __bf16* __restrict__ hi,
                                                   __bf16* __restrict__ lo, size_t idx) {
  __bf16 h = (__bf16)a;
  hi[idx] = h;
  lo[idx] = (__bf16)(a - (float)h);
}

// A-matrix fragment (16x32 bf16) from pre-split bf16, row-major, tile origin (row0,k0):
// lane<16 -> row=lane, K {0..7,16..23}; lane>=16 -> row=lane-16, K {8..15,24..31}.
static __device__ __forceinline__ v16bf load_a_bf16(const __bf16* __restrict__ A,
                                                    int lda, int lane) {
  const __bf16* p = A + (size_t)(lane & 15) * lda + ((lane & 16) ? 8 : 0);
  v8bf c0 = *(const v8bf*)p;         // 16B aligned
  v8bf c1 = *(const v8bf*)(p + 16);  // 16B aligned
  return __builtin_shufflevector(c0, c1, 0, 1, 2, 3, 4, 5, 6, 7,
                                 8, 9, 10, 11, 12, 13, 14, 15);
}

// B-matrix fragment (32x16 bf16) from a source whose rows are K-contiguous for a
// fixed output column (transposed weights Bt[n][k], or K-matrix rows).  At (n0,k0).
static __device__ __forceinline__ v16bf load_b_bf16(const __bf16* __restrict__ Bt,
                                                    int ldb, int lane) {
  const __bf16* p = Bt + (size_t)(lane & 15) * ldb + ((lane & 16) ? 8 : 0);
  v8bf c0 = *(const v8bf*)p;         // 16B aligned
  v8bf c1 = *(const v8bf*)(p + 16);  // 16B aligned
  return __builtin_shufflevector(c0, c1, 0, 1, 2, 3, 4, 5, 6, 7,
                                 8, 9, 10, 11, 12, 13, 14, 15);
}

// Fallback: in-register split of fp32 A fragment (used only by pv_slow path).
static __device__ __forceinline__ void split8(const float* __restrict__ p,
                                              v16bf& hi, v16bf& lo, int base) {
#pragma unroll
  for (int i = 0; i < 8; ++i) {
    float a = p[i];
    __bf16 h = (__bf16)a;
    hi[base + i] = h;
    lo[base + i] = (__bf16)(a - (float)h);
  }
}

static __device__ __forceinline__ FragPair load_a_f32(const float* __restrict__ A,
                                                      int lda, int lane) {
  FragPair f;
  const float* p = A + (size_t)(lane & 15) * lda + ((lane & 16) ? 8 : 0);
  split8(p,      f.hi, f.lo, 0);
  split8(p + 16, f.hi, f.lo, 8);
  return f;
}

// bf16x2 split product: c += hi*hi + hi*lo + lo*hi  (lo*lo below representable bits)
static __device__ __forceinline__ v8f wmma3(const FragPair& a, const FragPair& b, v8f c) {
  c = __builtin_amdgcn_wmma_f32_16x16x32_bf16(false, a.hi, false, b.hi, (short)0, c, false, false);
  c = __builtin_amdgcn_wmma_f32_16x16x32_bf16(false, a.hi, false, b.lo, (short)0, c, false, false);
  c = __builtin_amdgcn_wmma_f32_16x16x32_bf16(false, a.lo, false, b.hi, (short)0, c, false, false);
  return c;
}

static __device__ __forceinline__ float gelu_exact(float x) {
  return 0.5f * x * (1.0f + erff(x * 0.70710678118654752f));
}

// ------------------------------------------------------------ prep kernels --

// Wt[n][k] = split(W[k][n])
__global__ void prep_weight_kernel(const float* __restrict__ W,
                                   __bf16* __restrict__ Wt_hi,
                                   __bf16* __restrict__ Wt_lo) {
  int e = blockIdx.x * blockDim.x + threadIdx.x;   // e = n*512 + k (coalesced writes)
  int n = e >> 9, k = e & 511;
  split_store(W[(size_t)k * DM + n], Wt_hi, Wt_lo, (size_t)e);
}

// Elementwise fp32 -> bf16 hi/lo split
__global__ void split_kernel(const float* __restrict__ src,
                             __bf16* __restrict__ hi, __bf16* __restrict__ lo) {
  size_t i = (size_t)blockIdx.x * blockDim.x + threadIdx.x;
  split_store(src[i], hi, lo, i);
}

// vT[hb][d][k] = split(v[hb][k][d])
__global__ void transpose_split_v_kernel(const float* __restrict__ v,
                                         __bf16* __restrict__ vT_hi,
                                         __bf16* __restrict__ vT_lo) {
  __shared__ float tile[32][33];
  int hb = blockIdx.z;
  int kk0 = blockIdx.x * 32, d0 = blockIdx.y * 32;
  int t = threadIdx.x;
  int dx = t & 31, ky = t >> 5;
#pragma unroll
  for (int i = 0; i < 4; ++i) {
    int kk = ky + i * 8;
    tile[kk][dx] = v[((size_t)hb * SLEN + kk0 + kk) * DK + d0 + dx];
  }
  __syncthreads();
  int kx = t & 31, dy = t >> 5;
#pragma unroll
  for (int i = 0; i < 4; ++i) {
    int d = dy + i * 8;
    split_store(tile[kx][d], vT_hi, vT_lo,
                ((size_t)hb * DK + d0 + d) * SLEN + kk0 + kx);
  }
}

// --------------------------------------------------------- projection GEMM --
// C = GELU(A[M,512] @ W + bias); A pre-split bf16.  SPLIT_OUT: emit bf16 hi/lo
// (for q/k chains) instead of fp32.
// block = 256 threads (8 waves, 2x4); block tile 64(M) x 128(N); wave tile 32x32
template <bool SPLIT_OUT>
__global__ void proj_gemm_kernel(const __bf16* __restrict__ A_hi,
                                 const __bf16* __restrict__ A_lo,
                                 const __bf16* __restrict__ Wt_hi,
                                 const __bf16* __restrict__ Wt_lo,
                                 const float* __restrict__ bias,
                                 float* __restrict__ Cf,
                                 __bf16* __restrict__ C_hi,
                                 __bf16* __restrict__ C_lo) {
  int tid = threadIdx.x, lane = tid & 31, wave = tid >> 5;
  int wm = wave >> 2, wn = wave & 3;
  int m0 = blockIdx.y * 64 + wm * 32;
  int n0 = blockIdx.x * 128 + wn * 32;
  v8f acc[2][2] = {};
  for (int k0 = 0; k0 < DM; k0 += 32) {
    FragPair a0 = { load_a_bf16(A_hi + (size_t)m0 * DM + k0, DM, lane),
                    load_a_bf16(A_lo + (size_t)m0 * DM + k0, DM, lane) };
    FragPair a1 = { load_a_bf16(A_hi + (size_t)(m0 + 16) * DM + k0, DM, lane),
                    load_a_bf16(A_lo + (size_t)(m0 + 16) * DM + k0, DM, lane) };
    FragPair b0 = { load_b_bf16(Wt_hi + (size_t)n0 * DM + k0, DM, lane),
                    load_b_bf16(Wt_lo + (size_t)n0 * DM + k0, DM, lane) };
    FragPair b1 = { load_b_bf16(Wt_hi + (size_t)(n0 + 16) * DM + k0, DM, lane),
                    load_b_bf16(Wt_lo + (size_t)(n0 + 16) * DM + k0, DM, lane) };
    acc[0][0] = wmma3(a0, b0, acc[0][0]);
    acc[0][1] = wmma3(a0, b1, acc[0][1]);
    acc[1][0] = wmma3(a1, b0, acc[1][0]);
    acc[1][1] = wmma3(a1, b1, acc[1][1]);
  }
#pragma unroll
  for (int mi = 0; mi < 2; ++mi)
#pragma unroll
    for (int ni = 0; ni < 2; ++ni) {
      int col = n0 + ni * 16 + (lane & 15);
      float b = bias[col];
      int row = m0 + mi * 16 + ((lane & 16) ? 8 : 0);
#pragma unroll
      for (int g = 0; g < 8; ++g) {
        float val = gelu_exact(acc[mi][ni][g] + b);
        size_t o = (size_t)(row + g) * DM + col;
        if (SPLIT_OUT) split_store(val, C_hi, C_lo, o);
        else           Cf[o] = val;
      }
    }
}

// ------------------------------------------- scores: (Q @ K^T)*0.125 per hb --
__global__ void scores_kernel(const __bf16* __restrict__ q_hi,
                              const __bf16* __restrict__ q_lo,
                              const __bf16* __restrict__ k_hi,
                              const __bf16* __restrict__ k_lo,
                              float* __restrict__ attn) {
  int hb = blockIdx.z;
  const __bf16* Qh = q_hi + (size_t)hb * SLEN * DK;
  const __bf16* Ql = q_lo + (size_t)hb * SLEN * DK;
  const __bf16* Kh = k_hi + (size_t)hb * SLEN * DK;
  const __bf16* Kl = k_lo + (size_t)hb * SLEN * DK;
  float* Sm = attn + (size_t)hb * SLEN * SLEN;
  int tid = threadIdx.x, lane = tid & 31, wave = tid >> 5;
  int wm = wave >> 2, wn = wave & 3;
  int m0 = blockIdx.y * 64 + wm * 32;
  int n0 = blockIdx.x * 128 + wn * 32;
  v8f acc[2][2] = {};
#pragma unroll
  for (int k0 = 0; k0 < DK; k0 += 32) {
    FragPair a0 = { load_a_bf16(Qh + (size_t)m0 * DK + k0, DK, lane),
                    load_a_bf16(Ql + (size_t)m0 * DK + k0, DK, lane) };
    FragPair a1 = { load_a_bf16(Qh + (size_t)(m0 + 16) * DK + k0, DK, lane),
                    load_a_bf16(Ql + (size_t)(m0 + 16) * DK + k0, DK, lane) };
    FragPair b0 = { load_b_bf16(Kh + (size_t)n0 * DK + k0, DK, lane),
                    load_b_bf16(Kl + (size_t)n0 * DK + k0, DK, lane) };
    FragPair b1 = { load_b_bf16(Kh + (size_t)(n0 + 16) * DK + k0, DK, lane),
                    load_b_bf16(Kl + (size_t)(n0 + 16) * DK + k0, DK, lane) };
    acc[0][0] = wmma3(a0, b0, acc[0][0]);
    acc[0][1] = wmma3(a0, b1, acc[0][1]);
    acc[1][0] = wmma3(a1, b0, acc[1][0]);
    acc[1][1] = wmma3(a1, b1, acc[1][1]);
  }
#pragma unroll
  for (int mi = 0; mi < 2; ++mi)
#pragma unroll
    for (int ni = 0; ni < 2; ++ni) {
      int col = n0 + ni * 16 + (lane & 15);
      int row = m0 + mi * 16 + ((lane & 16) ? 8 : 0);
#pragma unroll
      for (int g = 0; g < 8; ++g)
        Sm[(size_t)(row + g) * SLEN + col] = acc[mi][ni][g] * 0.125f;
    }
}

// -------------------------- row softmax; optionally emit bf16 hi/lo copies --
template <bool EMIT_SPLIT>
__global__ void softmax_kernel(float* __restrict__ attn,
                               __bf16* __restrict__ a_hi,
                               __bf16* __restrict__ a_lo) {
  __shared__ float red[256];
  size_t row = blockIdx.x;
  float* p = attn + row * (size_t)SLEN;
  int t = threadIdx.x;
  float v[8];
  float mx = -INFINITY;
#pragma unroll
  for (int i = 0; i < 8; ++i) { v[i] = p[t + i * 256]; mx = fmaxf(mx, v[i]); }
  red[t] = mx; __syncthreads();
  for (int s = 128; s > 0; s >>= 1) { if (t < s) red[t] = fmaxf(red[t], red[t + s]); __syncthreads(); }
  float rowmax = red[0];
  __syncthreads();
  float sum = 0.0f;
#pragma unroll
  for (int i = 0; i < 8; ++i) { v[i] = expf(v[i] - rowmax); sum += v[i]; }
  red[t] = sum; __syncthreads();
  for (int s = 128; s > 0; s >>= 1) { if (t < s) red[t] += red[t + s]; __syncthreads(); }
  float inv = 1.0f / red[0];
#pragma unroll
  for (int i = 0; i < 8; ++i) {
    float val = v[i] * inv;
    size_t o = row * (size_t)SLEN + t + i * 256;
    p[t + i * 256] = val;
    if (EMIT_SPLIT) split_store(val, a_hi, a_lo, o);
  }
}

// --------------------------------------------- PV: O = attn @ V per (h,b) --
// block 256 = 8 waves (4 M x 2 N); block tile 128(M) x 64(N); result pre-split.
__global__ void pv_fast_kernel(const __bf16* __restrict__ a_hi,
                               const __bf16* __restrict__ a_lo,
                               const __bf16* __restrict__ vT_hi,
                               const __bf16* __restrict__ vT_lo,
                               __bf16* __restrict__ O_hi,
                               __bf16* __restrict__ O_lo) {
  int hb = blockIdx.z;
  const __bf16* Ph = a_hi + (size_t)hb * SLEN * SLEN;
  const __bf16* Pl = a_lo + (size_t)hb * SLEN * SLEN;
  const __bf16* Bh = vT_hi + (size_t)hb * DK * SLEN;
  const __bf16* Bl = vT_lo + (size_t)hb * DK * SLEN;
  int tid = threadIdx.x, lane = tid & 31, wave = tid >> 5;
  int wm = wave >> 1, wn = wave & 1;
  int m0 = blockIdx.y * 128 + wm * 32;
  int n0 = wn * 32;
  v8f acc[2][2] = {};
  for (int k0 = 0; k0 < SLEN; k0 += 32) {
    FragPair a0 = { load_a_bf16(Ph + (size_t)m0 * SLEN + k0, SLEN, lane),
                    load_a_bf16(Pl + (size_t)m0 * SLEN + k0, SLEN, lane) };
    FragPair a1 = { load_a_bf16(Ph + (size_t)(m0 + 16) * SLEN + k0, SLEN, lane),
                    load_a_bf16(Pl + (size_t)(m0 + 16) * SLEN + k0, SLEN, lane) };
    FragPair b0 = { load_b_bf16(Bh + (size_t)n0 * SLEN + k0, SLEN, lane),
                    load_b_bf16(Bl + (size_t)n0 * SLEN + k0, SLEN, lane) };
    FragPair b1 = { load_b_bf16(Bh + (size_t)(n0 + 16) * SLEN + k0, SLEN, lane),
                    load_b_bf16(Bl + (size_t)(n0 + 16) * SLEN + k0, SLEN, lane) };
    acc[0][0] = wmma3(a0, b0, acc[0][0]);
    acc[0][1] = wmma3(a0, b1, acc[0][1]);
    acc[1][0] = wmma3(a1, b0, acc[1][0]);
    acc[1][1] = wmma3(a1, b1, acc[1][1]);
  }
  size_t obase = (size_t)hb * SLEN * DK;
#pragma unroll
  for (int mi = 0; mi < 2; ++mi)
#pragma unroll
    for (int ni = 0; ni < 2; ++ni) {
      int col = n0 + ni * 16 + (lane & 15);
      int row = m0 + mi * 16 + ((lane & 16) ? 8 : 0);
#pragma unroll
      for (int g = 0; g < 8; ++g)
        split_store(acc[mi][ni][g], O_hi, O_lo, obase + (size_t)(row + g) * DK + col);
    }
}

// Fallback PV: attn stays fp32 and is split in-register (small-workspace path).
__global__ void pv_slow_kernel(const float* __restrict__ attn,
                               const __bf16* __restrict__ vT_hi,
                               const __bf16* __restrict__ vT_lo,
                               __bf16* __restrict__ O_hi,
                               __bf16* __restrict__ O_lo) {
  int hb = blockIdx.z;
  const float* P = attn + (size_t)hb * SLEN * SLEN;
  const __bf16* Bh = vT_hi + (size_t)hb * DK * SLEN;
  const __bf16* Bl = vT_lo + (size_t)hb * DK * SLEN;
  int tid = threadIdx.x, lane = tid & 31, wave = tid >> 5;
  int wm = wave >> 1, wn = wave & 1;
  int m0 = blockIdx.y * 128 + wm * 32;
  int n0 = wn * 32;
  v8f acc[2][2] = {};
  for (int k0 = 0; k0 < SLEN; k0 += 32) {
    FragPair a0 = load_a_f32(P + (size_t)m0 * SLEN + k0, SLEN, lane);
    FragPair a1 = load_a_f32(P + (size_t)(m0 + 16) * SLEN + k0, SLEN, lane);
    FragPair b0 = { load_b_bf16(Bh + (size_t)n0 * SLEN + k0, SLEN, lane),
                    load_b_bf16(Bl + (size_t)n0 * SLEN + k0, SLEN, lane) };
    FragPair b1 = { load_b_bf16(Bh + (size_t)(n0 + 16) * SLEN + k0, SLEN, lane),
                    load_b_bf16(Bl + (size_t)(n0 + 16) * SLEN + k0, SLEN, lane) };
    acc[0][0] = wmma3(a0, b0, acc[0][0]);
    acc[0][1] = wmma3(a0, b1, acc[0][1]);
    acc[1][0] = wmma3(a1, b0, acc[1][0]);
    acc[1][1] = wmma3(a1, b1, acc[1][1]);
  }
  size_t obase = (size_t)hb * SLEN * DK;
#pragma unroll
  for (int mi = 0; mi < 2; ++mi)
#pragma unroll
    for (int ni = 0; ni < 2; ++ni) {
      int col = n0 + ni * 16 + (lane & 15);
      int row = m0 + mi * 16 + ((lane & 16) ? 8 : 0);
#pragma unroll
      for (int g = 0; g < 8; ++g)
        split_store(acc[mi][ni][g], O_hi, O_lo, obase + (size_t)(row + g) * DK + col);
    }
}

// ---------------------------------------------------------------- launcher --

extern "C" void kernel_launch(void* const* d_in, const int* in_sizes, int n_in,
                              void* d_out, int out_size, void* d_ws, size_t ws_size,
                              hipStream_t stream) {
  (void)in_sizes; (void)n_in; (void)out_size;
  const float* x   = (const float*)d_in[0];
  const float* w_q = (const float*)d_in[1];
  const float* b_q = (const float*)d_in[2];
  const float* w_k = (const float*)d_in[3];
  const float* b_k = (const float*)d_in[4];
  const float* w_v = (const float*)d_in[5];
  const float* b_v = (const float*)d_in[6];
  const float* w_o = (const float*)d_in[7];
  const float* b_o = (const float*)d_in[8];

  float* out  = (float*)d_out;                       // [B,S,D] fp32
  float* attn = out + (size_t)MROWS * DM;            // [H,B,S,S] fp32

  const size_t NE = (size_t)MROWS * DM;              // 2,097,152 (also NHB*DK*SLEN)
  const size_t NA = (size_t)NHB * SLEN * SLEN;       // 67,108,864

  // workspace carve-up (all segment sizes are multiples of 4 MB -> aligned)
  __bf16* x_hi  = (__bf16*)d_ws;
  __bf16* x_lo  = x_hi + NE;
  __bf16* q_hi  = x_lo + NE;
  __bf16* q_lo  = q_hi + NE;
  __bf16* k_hi  = q_lo + NE;
  __bf16* k_lo  = k_hi + NE;
  float*  vv    = (float*)(k_lo + NE);
  __bf16* vT_hi = (__bf16*)(vv + NE);
  __bf16* vT_lo = vT_hi + NE;
  __bf16* oa_hi = vT_lo + NE;
  __bf16* oa_lo = oa_hi + NE;
  __bf16* wbase = oa_lo + NE;                        // 8 x DM*DM bf16
  __bf16* wq_hi = wbase + 0 * (size_t)DM * DM;
  __bf16* wq_lo = wbase + 1 * (size_t)DM * DM;
  __bf16* wk_hi = wbase + 2 * (size_t)DM * DM;
  __bf16* wk_lo = wbase + 3 * (size_t)DM * DM;
  __bf16* wv_hi = wbase + 4 * (size_t)DM * DM;
  __bf16* wv_lo = wbase + 5 * (size_t)DM * DM;
  __bf16* wo_hi = wbase + 6 * (size_t)DM * DM;
  __bf16* wo_lo = wbase + 7 * (size_t)DM * DM;
  __bf16* a_hi  = wbase + 8 * (size_t)DM * DM;       // attn splits (fast path only)
  __bf16* a_lo  = a_hi + NA;

  const size_t need_fast = (size_t)((char*)(a_lo + NA) - (char*)d_ws);
  const bool fast = ws_size >= need_fast;

  dim3 blk(256);

  prep_weight_kernel<<<DM * DM / 256, blk, 0, stream>>>(w_q, wq_hi, wq_lo);
  prep_weight_kernel<<<DM * DM / 256, blk, 0, stream>>>(w_k, wk_hi, wk_lo);
  prep_weight_kernel<<<DM * DM / 256, blk, 0, stream>>>(w_v, wv_hi, wv_lo);
  prep_weight_kernel<<<DM * DM / 256, blk, 0, stream>>>(w_o, wo_hi, wo_lo);

  split_kernel<<<NE / 256, blk, 0, stream>>>(x, x_hi, x_lo);

  dim3 pgrid(DM / 128, MROWS / 64);
  proj_gemm_kernel<true ><<<pgrid, blk, 0, stream>>>(x_hi, x_lo, wq_hi, wq_lo, b_q,
                                                     nullptr, q_hi, q_lo);
  proj_gemm_kernel<true ><<<pgrid, blk, 0, stream>>>(x_hi, x_lo, wk_hi, wk_lo, b_k,
                                                     nullptr, k_hi, k_lo);
  proj_gemm_kernel<false><<<pgrid, blk, 0, stream>>>(x_hi, x_lo, wv_hi, wv_lo, b_v,
                                                     vv, nullptr, nullptr);

  transpose_split_v_kernel<<<dim3(SLEN / 32, DK / 32, NHB), blk, 0, stream>>>(vv, vT_hi, vT_lo);

  scores_kernel<<<dim3(SLEN / 128, SLEN / 64, NHB), blk, 0, stream>>>(q_hi, q_lo, k_hi, k_lo, attn);

  if (fast) {
    softmax_kernel<true ><<<NHB * SLEN, blk, 0, stream>>>(attn, a_hi, a_lo);
    pv_fast_kernel<<<dim3(1, SLEN / 128, NHB), blk, 0, stream>>>(a_hi, a_lo, vT_hi, vT_lo,
                                                                 oa_hi, oa_lo);
  } else {
    softmax_kernel<false><<<NHB * SLEN, blk, 0, stream>>>(attn, nullptr, nullptr);
    pv_slow_kernel<<<dim3(1, SLEN / 128, NHB), blk, 0, stream>>>(attn, vT_hi, vT_lo,
                                                                 oa_hi, oa_lo);
  }

  proj_gemm_kernel<false><<<pgrid, blk, 0, stream>>>(oa_hi, oa_lo, wo_hi, wo_lo, b_o,
                                                     out, nullptr, nullptr);
}
